// PointTransformerSlim_53755810677212
// MI455X (gfx1250) — compile-verified, matrix operand order checked by reading, not run on hardware
//
#include <hip/hip_runtime.h>
#include <hip/hip_bf16.h>

typedef __attribute__((ext_vector_type(16))) _Float16 v16h;
typedef __attribute__((ext_vector_type(8)))  _Float16 v8h;
typedef __attribute__((ext_vector_type(8)))  float    v8f;
typedef __attribute__((ext_vector_type(4)))  unsigned v4u;
typedef __attribute__((ext_vector_type(8)))  int      v8i;
typedef __attribute__((ext_vector_type(4)))  int      v4i;

constexpr int CD  = 128;   // channels
constexpr int PN  = 256;   // points
constexpr int NB  = 8;     // batch
constexpr int STH = 136;   // f16 LDS row stride (16B-aligned, padded)

#if defined(__has_builtin)
#if __has_builtin(__builtin_amdgcn_tensor_load_to_lds) && __has_builtin(__builtin_amdgcn_s_wait_tensorcnt)
#define USE_TDM 1
#endif
#endif
#ifndef USE_TDM
#define USE_TDM 0
#endif

// ---------------------------------------------------------------------------
// TDM: 1-D contiguous global -> LDS DMA with optional LDS row padding.
// D# per CDNA5 ISA:
//   group0: count=1 | lds_addr | global_addr[56:0] | type=2
//   group1: data_size, pad ctl, tensor_dim0=tile_dim0=nelem, dim1=1, stride=nelem
// pad_interval code: insert pad after 2^(code+1) DWORDs; pad_amount code: (code+1) DWORDs.
// ---------------------------------------------------------------------------
#if USE_TDM
__device__ __forceinline__ void tdm_load_1d(unsigned lds_byte_off, const void* gptr,
                                            unsigned nelem, unsigned dsz_code,
                                            unsigned pad_en, unsigned pad_interval,
                                            unsigned pad_amount) {
  unsigned long long ga = (unsigned long long)(uintptr_t)gptr;
  v4u g0;
  g0[0] = 1u;                                               // count=1, user descriptor
  g0[1] = lds_byte_off;                                     // lds_addr
  g0[2] = (unsigned)(ga & 0xFFFFFFFFu);                     // global_addr[31:0]
  g0[3] = (unsigned)((ga >> 32) & 0x1FFFFFFu) | (2u << 30); // global_addr[56:32] | type=2
  v8i g1;
  unsigned w0 = (dsz_code << 16) | (pad_en << 20) | (pad_interval << 22) | (pad_amount << 25);
  g1[0] = (int)w0;
  g1[1] = (int)((nelem & 0xFFFFu) << 16);                   // tensor_dim0[15:0]  @ bits 63:48
  g1[2] = (int)(((nelem >> 16) & 0xFFFFu) | (1u << 16));    // tensor_dim0[31:16] | tensor_dim1=1
  g1[3] = (int)((nelem & 0xFFFFu) << 16);                   // tile_dim0 @ bits 127:112
  g1[4] = 0;                                                // tile_dim1=0, tile_dim2=0
  g1[5] = (int)nelem;                                       // tensor_dim0_stride[31:0]
  g1[6] = 0;
  g1[7] = 0;
  v4i z4 = {0, 0, 0, 0};
#if __clang_major__ >= 23
  v8i z8 = {0, 0, 0, 0, 0, 0, 0, 0};
  __builtin_amdgcn_tensor_load_to_lds(g0, g1, z4, z4, z8, 0);
#else
  __builtin_amdgcn_tensor_load_to_lds(g0, g1, z4, z4, 0);
#endif
}
__device__ __forceinline__ unsigned lds_off(const void* p) {
  return (unsigned)(uintptr_t)p;   // low 32 bits of generic addr = LDS byte offset
}
#endif

// ---------------------------------------------------------------------------
// K1a: fold weights.  WqW1 = Wq@W1, WkW1 = Wk@W1 (f32), WpW1T[j][c] = (Wp2@W1)[c][j]
// as f16 (B-fragment-friendly transposed), biasJ = b1 + bp2@W1.
// ---------------------------------------------------------------------------
__global__ void k_fold_weights(const float* __restrict__ Wq, const float* __restrict__ Wk,
                               const float* __restrict__ W1, const float* __restrict__ Wp2,
                               const float* __restrict__ b1, const float* __restrict__ bp2,
                               float* __restrict__ WqW1, float* __restrict__ WkW1,
                               _Float16* __restrict__ WpW1T, float* __restrict__ biasJ) {
  int c = blockIdx.x;      // 128 blocks
  int j = threadIdx.x;     // 128 threads
  float sq = 0.f, sk = 0.f, sp = 0.f;
  for (int m = 0; m < CD; ++m) {
    float w1 = W1[m * CD + j];
    sq = fmaf(Wq[c * CD + m], w1, sq);
    sk = fmaf(Wk[c * CD + m], w1, sk);
    sp = fmaf(Wp2[c * CD + m], w1, sp);
  }
  WqW1[c * CD + j] = sq;
  WkW1[c * CD + j] = sk;
  WpW1T[j * CD + c] = (_Float16)sp;
  if (c == 0) {
    float s = b1[j];
    for (int m = 0; m < CD; ++m) s = fmaf(bp2[m], W1[m * CD + j], s);
    biasJ[j] = s;
  }
}

// ---------------------------------------------------------------------------
// Fragment loader: v16h operand from LDS-resident f16 matrix, CDNA5 16-bit
// 16x32 per-lane layout (lanes 0-15: K chunks +0/+16, lanes 16-31: +8/+24).
// ---------------------------------------------------------------------------
__device__ __forceinline__ v16h load_frag(const _Float16* base, int rc0, int k0,
                                          int lane, int stride) {
  int half = lane >> 4;
  int idx  = lane & 15;
  const _Float16* p = base + (rc0 + idx) * stride + k0 + half * 8;
  v8h lo = *(const v8h*)(p);
  v8h hi = *(const v8h*)(p + 16);
  return __builtin_shufflevector(lo, hi, 0,1,2,3,4,5,6,7,8,9,10,11,12,13,14,15);
}

// ---------------------------------------------------------------------------
// K1b: C[2048,128] = A[2048,128] @ B[128,128], f32 in/out, f16 WMMA inside.
// A fragments hoisted and reused across all 8 N-tiles.
// ---------------------------------------------------------------------------
__global__ void k_gemm_x(const float* __restrict__ A, const float* __restrict__ B,
                         float* __restrict__ Cg) {
  __shared__ __align__(16) _Float16 sA[128 * STH];
  __shared__ __align__(16) _Float16 sB[128 * STH];   // sB[j][c] = B[c][j]
  int tid = threadIdx.x, lane = tid & 31, wave = tid >> 5;
  int row0 = blockIdx.x * 128;

  for (int i = tid; i < 128 * 128; i += 256) {
    int r = i >> 7, c = i & 127;
    sA[r * STH + c] = (_Float16)A[(row0 + r) * CD + c];
    sB[r * STH + c] = (_Float16)B[c * CD + r];
  }
  __syncthreads();

  int mt = wave;
  v16h afr[4];
#pragma unroll
  for (int k = 0; k < 4; ++k) afr[k] = load_frag(sA, mt * 16, k * 32, lane, STH);

  for (int jt = 0; jt < 8; ++jt) {
    v8f acc = {};
#pragma unroll
    for (int k = 0; k < 4; ++k) {
      v16h b = load_frag(sB, jt * 16, k * 32, lane, STH);
      acc = __builtin_amdgcn_wmma_f32_16x16x32_f16(false, afr[k], false, b,
                                                   (short)0, acc, false, false);
    }
    int col = jt * 16 + (lane & 15);
    int rbase = row0 + mt * 16 + ((lane >> 4) << 3);
#pragma unroll
    for (int r = 0; r < 8; ++r)
      Cg[(rbase + r) * CD + col] = acc[r];
  }
}

// ---------------------------------------------------------------------------
// K2: fused attention.  One workgroup = (n, 16 consecutive a).
// TDM stages WpW1T (with hardware row-padding), kW1 tile, and double-buffered
// qW1 tiles; WMMA computes h1 @ WpW1T; epilogue does relu-dot with W2;
// softmax over b; att@v + residual.
// ---------------------------------------------------------------------------
__global__ void k_attention(const float* __restrict__ x, const float* __restrict__ pos,
                            const float* __restrict__ qW1, const float* __restrict__ kW1,
                            const float* __restrict__ vbuf, const _Float16* __restrict__ WpW1T,
                            const float* __restrict__ biasJ, const float* __restrict__ W2,
                            const float* __restrict__ b2, const float* __restrict__ Wp1,
                            const float* __restrict__ bp1, float* __restrict__ out) {
  __shared__ __align__(16) _Float16 sWp[128 * STH];  // folded weight, B-transposed, padded
  __shared__ __align__(16) _Float16 sH1[256 * STH];  // h1 tile: 256 (a,b) rows x 128
  __shared__ __align__(16) float sQ[2][16 * 128];    // double-buffered qW1 tiles
  __shared__ __align__(16) float sK[16 * 128];
  __shared__ float sLg[16 * 260];                    // logits [a_local][b]
  __shared__ float sBias[128], sW2[128], sP0[128], sP1[128], sBp[128];

  int tid = threadIdx.x, lane = tid & 31, wave = tid >> 5;
  int n  = blockIdx.x >> 4;
  int a0 = (blockIdx.x & 15) << 4;

  const float* qbase = qW1 + (size_t)(n * PN) * CD;

#if USE_TDM
  if (wave == 0) {
    // WpW1T: 16384 f16, pad 4 DWORDs (8 f16) after every 64 DWORDs (128 f16):
    // pad_interval code 5 (64 DWORDs), pad_amount code 3 (4 DWORDs).
    tdm_load_1d(lds_off(sWp), WpW1T, 128 * 128, /*dsz f16*/1, 1, 5, 3);
    // kW1 a-tile: 2048 contiguous f32, unpadded.
    tdm_load_1d(lds_off(sK), kW1 + (size_t)(n * PN + a0) * CD, 16 * 128, /*f32*/2, 0, 0, 0);
    // first qW1 b-tile into buffer 0.
    tdm_load_1d(lds_off(sQ[0]), qbase, 16 * 128, 2, 0, 0, 0);
  }
#else
  for (int i = tid; i < 128 * 128; i += 256) {
    int j = i >> 7, c = i & 127;
    sWp[j * STH + c] = WpW1T[i];
  }
  for (int i = tid; i < 16 * 128; i += 256)
    sK[i] = kW1[(size_t)(n * PN + a0) * CD + i];
  for (int i = tid; i < 16 * 128; i += 256)
    sQ[0][i] = qbase[i];
#endif
  if (tid < 128) {
    sBias[tid] = biasJ[tid];
    sW2[tid]   = W2[tid];
    sP0[tid]   = Wp1[tid];
    sP1[tid]   = Wp1[128 + tid];
    sBp[tid]   = bp1[tid];
  }
  float bias2 = b2[0];

  for (int bt = 0; bt < 16; ++bt) {
    int b0 = bt << 4;
    int cur = bt & 1;
#if USE_TDM
    if (wave == 0 && bt < 15)   // prefetch next qW1 tile into the other buffer
      tdm_load_1d(lds_off(sQ[cur ^ 1]), qbase + (size_t)(b0 + 16) * CD, 16 * 128, 2, 0, 0, 0);
#else
    if (bt > 0) {
      for (int i = tid; i < 16 * 128; i += 256)
        sQ[cur][i] = qbase[(size_t)b0 * CD + i];
    }
#endif
    // h1 tile: one (a_local,b_local) row per thread
    {
      int al = tid >> 4, bl = tid & 15;
      const float* pp = pos + ((((size_t)n * PN + (a0 + al)) * PN + (b0 + bl)) << 1);
      float px = pp[0], py = pp[1];
      _Float16* hrow = sH1 + tid * STH;
#pragma unroll 4
      for (int c = 0; c < 128; ++c) {
        float h = fmaf(px, sP0[c], fmaf(py, sP1[c], sBp[c]));
        hrow[c] = (_Float16)fmaxf(h, 0.f);
      }
    }
#if USE_TDM
    if (wave == 0) {
      // current tile (and, at bt==0, sWp/sK) complete; next tile stays in flight.
      // SIMM16 must be a literal constant.
      if (bt < 15) __builtin_amdgcn_s_wait_tensorcnt(1);
      else         __builtin_amdgcn_s_wait_tensorcnt(0);
    }
#endif
    __syncthreads();

    const float* q = sQ[cur];
    // each wave owns 2 M-tiles of the 256-row tile
    for (int t = 0; t < 2; ++t) {
      int mt = wave * 2 + t;              // a_local == mt
      v16h afr[4];
#pragma unroll
      for (int k = 0; k < 4; ++k) afr[k] = load_frag(sH1, mt * 16, k * 32, lane, STH);

      float preg[8];
#pragma unroll
      for (int r = 0; r < 8; ++r) preg[r] = 0.f;

      for (int jt = 0; jt < 8; ++jt) {
        v8f acc = {};
#pragma unroll
        for (int k = 0; k < 4; ++k) {
          v16h b = load_frag(sWp, jt * 16, k * 32, lane, STH);
          acc = __builtin_amdgcn_wmma_f32_16x16x32_f16(false, afr[k], false, b,
                                                       (short)0, acc, false, false);
        }
        int j  = jt * 16 + (lane & 15);
        float kb = sBias[j] - sK[mt * 128 + j];
        float w2 = sW2[j];
        int rb = (lane >> 4) << 3;
#pragma unroll
        for (int r = 0; r < 8; ++r) {
          float tv = acc[r] + q[(rb + r) * 128 + j] + kb;
          preg[r] += fmaxf(tv, 0.f) * w2;
        }
      }
#pragma unroll
      for (int r = 0; r < 8; ++r) {
        float v = preg[r];
        v += __shfl_xor(v, 1, 16);
        v += __shfl_xor(v, 2, 16);
        v += __shfl_xor(v, 4, 16);
        v += __shfl_xor(v, 8, 16);
        preg[r] = v;
      }
      if ((lane & 15) == 0) {
        int rb = (lane >> 4) << 3;
#pragma unroll
        for (int r = 0; r < 8; ++r)
          sLg[mt * 260 + b0 + rb + r] = preg[r] + bias2;
      }
    }
    __syncthreads();
  }

  // softmax over b for each a_local (16 threads per row)
  {
    int row = tid >> 4, sub = tid & 15;
    float* L = sLg + row * 260;
    float m = -1e30f;
    for (int b = sub; b < 256; b += 16) m = fmaxf(m, L[b]);
    m = fmaxf(m, __shfl_xor(m, 1, 16));
    m = fmaxf(m, __shfl_xor(m, 2, 16));
    m = fmaxf(m, __shfl_xor(m, 4, 16));
    m = fmaxf(m, __shfl_xor(m, 8, 16));
    float s = 0.f;
    for (int b = sub; b < 256; b += 16) { float e = __expf(L[b] - m); L[b] = e; s += e; }
    s += __shfl_xor(s, 1, 16);
    s += __shfl_xor(s, 2, 16);
    s += __shfl_xor(s, 4, 16);
    s += __shfl_xor(s, 8, 16);
    float inv = 1.f / s;
    for (int b = sub; b < 256; b += 16) L[b] *= inv;
  }
  __syncthreads();

  // fuse: out[n, a0+al, c] = x + sum_b att[al][b] * v[n,b,c]
  {
    int al0 = (tid >> 7) << 3;   // 0 or 8
    int c = tid & 127;
    const float* vb = vbuf + ((size_t)n * PN) * CD + c;
#pragma unroll
    for (int i = 0; i < 8; ++i) {
      int al = al0 + i;
      const float* L = sLg + al * 260;
      float s = 0.f;
      for (int b = 0; b < 256; ++b) s = fmaf(L[b], vb[b * CD], s);
      size_t o = ((size_t)(n * PN) + a0 + al) * CD + c;
      out[o] = x[o] + s;
    }
  }
}

// ---------------------------------------------------------------------------
extern "C" void kernel_launch(void* const* d_in, const int* in_sizes, int n_in,
                              void* d_out, int out_size, void* d_ws, size_t ws_size,
                              hipStream_t stream) {
  const float* x   = (const float*)d_in[0];
  const float* pos = (const float*)d_in[1];
  const float* Wq  = (const float*)d_in[2];
  const float* Wk  = (const float*)d_in[3];
  const float* Wv  = (const float*)d_in[4];
  const float* W1  = (const float*)d_in[5];
  const float* b1  = (const float*)d_in[6];
  const float* W2  = (const float*)d_in[7];
  const float* b2  = (const float*)d_in[8];
  const float* Wp1 = (const float*)d_in[9];
  const float* bp1 = (const float*)d_in[10];
  const float* Wp2 = (const float*)d_in[11];
  const float* bp2 = (const float*)d_in[12];
  float* out = (float*)d_out;

  char* ws = (char*)d_ws;
  float*    WqW1  = (float*)(ws + 0);
  float*    WkW1  = (float*)(ws + 65536);
  _Float16* WpW1T = (_Float16*)(ws + 131072);
  float*    biasJ = (float*)(ws + 163840);
  float*    qW1   = (float*)(ws + 262144);
  float*    kW1   = (float*)(ws + 262144 + 1048576);
  float*    vb    = (float*)(ws + 262144 + 2 * 1048576);

  k_fold_weights<<<128, 128, 0, stream>>>(Wq, Wk, W1, Wp2, b1, bp2,
                                          WqW1, WkW1, WpW1T, biasJ);
  k_gemm_x<<<16, 256, 0, stream>>>(x, WqW1, qW1);
  k_gemm_x<<<16, 256, 0, stream>>>(x, WkW1, kW1);
  k_gemm_x<<<16, 256, 0, stream>>>(x, Wv, vb);
  k_attention<<<NB * 16, 256, 0, stream>>>(x, pos, qW1, kW1, vb, WpW1T,
                                           biasJ, W2, b2, Wp1, bp1, out);
}
